// MultiheadAttention_67516885893150
// MI455X (gfx1250) — matrix operand order repaired and checked by reading, not compiled, not run on hardware
//
#include <hip/hip_runtime.h>

#define S_LEN   2048
#define DMODEL  512
#define NHEADS  8
#define DK      64
#define BATCH   4
#define MROWS   (BATCH * S_LEN)   /* 8192 */
#define SC_LD   2056              /* 2048 + 8 pad: conflict-free LDS rows */

typedef __attribute__((ext_vector_type(16))) __bf16 bf16x16;
typedef __attribute__((ext_vector_type(8)))  __bf16 bf16x8;
typedef __attribute__((ext_vector_type(8)))  float  f32x8;

// ---------------------------------------------------------------------------
// WMMA helpers (CDNA5 V_WMMA_F32_16X16X32_BF16, wave32)
// ---------------------------------------------------------------------------
__device__ __forceinline__ f32x8 wmma_bf16(bf16x16 a, bf16x16 b, f32x8 c) {
  // (neg_a, A, neg_b, B, c_mod, C, reuse_a, reuse_b)
  return __builtin_amdgcn_wmma_f32_16x16x32_bf16(false, a, false, b, (short)0, c,
                                                 false, false);
}

// A-matrix 16x32 bf16 fragment from a row-major [*, ld] matrix.
// ISA 7.12.2: lane L holds row M=L&15; lanes 0-15: K={0..7,16..23},
// lanes 16-31: K={8..15,24..31}; element j<8 -> K=kOff+j, j>=8 -> K=kOff+16+j-8.
__device__ __forceinline__ bf16x16 load_a_frag(const __bf16* base, int ld,
                                               int mBase, int kBase, int lane) {
  int m    = mBase + (lane & 15);
  int kOff = (lane & 16) ? 8 : 0;
  const __bf16* p = base + (size_t)m * ld + kBase + kOff;
  union { bf16x16 v; bf16x8 h[2]; } u;
  u.h[0] = *(const bf16x8*)(p);        // K = kOff + 0..7
  u.h[1] = *(const bf16x8*)(p + 16);   // K = kOff + 16..23
  return u.v;
}

// B-matrix 32x16 bf16 fragment where the source is stored as Bt[n][c]
// (row-major over the contraction axis c).  j<8: b[j] = Bt[nBase+(L&15)][cBase+cOff+j],
// j>=8: c = cBase+cOff+16+j-8; cOff = 8 for lanes 16-31 (same K order as A).
__device__ __forceinline__ bf16x16 load_bt_frag(const __bf16* bt, int ld,
                                                int nBase, int cBase, int lane) {
  int n    = nBase + (lane & 15);
  int cOff = (lane & 16) ? 8 : 0;
  const __bf16* p = bt + (size_t)n * ld + cBase + cOff;
  union { bf16x16 v; bf16x8 h[2]; } u;
  u.h[0] = *(const bf16x8*)(p);
  u.h[1] = *(const bf16x8*)(p + 16);
  return u.v;
}

// ---------------------------------------------------------------------------
// fp32 -> bf16 conversion, 8 elements / thread
// ---------------------------------------------------------------------------
__global__ void cvt_f32_bf16(const float* __restrict__ src,
                             __bf16* __restrict__ dst, int n8) {
  int i = blockIdx.x * blockDim.x + threadIdx.x;
  if (i >= n8) return;
  const float4* s = (const float4*)src + (size_t)i * 2;
  float4 x0 = s[0], x1 = s[1];
  bf16x8 o;
  o[0] = (__bf16)x0.x; o[1] = (__bf16)x0.y; o[2] = (__bf16)x0.z; o[3] = (__bf16)x0.w;
  o[4] = (__bf16)x1.x; o[5] = (__bf16)x1.y; o[6] = (__bf16)x1.z; o[7] = (__bf16)x1.w;
  *(bf16x8*)(dst + (size_t)i * 8) = o;
}

// ---------------------------------------------------------------------------
// y = X @ W^T   X:[8192,512] bf16 row-major, W:[512,512] bf16 row-major
// (W row-major IS Bt for x@W^T).  One wave per 16x16 tile, 4 waves / block.
// mode 0: bf16 out, layout [B,H,S,DK]   (Q, K heads)
// mode 1: bf16 out, layout [B,H,DK,S]   (V transposed for PV B-frags)
// mode 2: fp32 out, row-major [8192,512] (final projection -> d_out)
// ---------------------------------------------------------------------------
__global__ void proj_gemm(const __bf16* __restrict__ X,
                          const __bf16* __restrict__ W,
                          void* __restrict__ dstv, int mode) {
  int lane  = threadIdx.x & 31;
  int w     = threadIdx.x >> 5;
  int mBase = blockIdx.x * 16;
  int nBase = blockIdx.y * 64 + w * 16;

  f32x8 acc = {};
#pragma unroll
  for (int kb = 0; kb < DMODEL; kb += 32) {
    bf16x16 a = load_a_frag(X, DMODEL, mBase, kb, lane);
    bf16x16 b = load_bt_frag(W, DMODEL, nBase, kb, lane);
    acc = wmma_bf16(a, b, acc);
  }

  int rbase = (lane & 16) ? 8 : 0;
  int n     = nBase + (lane & 15);
  int h     = n >> 6;        // head
  int dk    = n & 63;        // dim within head
#pragma unroll
  for (int r = 0; r < 8; ++r) {
    int m = mBase + rbase + r;
    int b_ = m >> 11;        // batch  (S = 2048)
    int s_ = m & 2047;       // sequence position
    if (mode == 0) {
      ((__bf16*)dstv)[(((size_t)(b_ * NHEADS + h) * S_LEN) + s_) * DK + dk] =
          (__bf16)acc[r];
    } else if (mode == 1) {
      ((__bf16*)dstv)[(((size_t)(b_ * NHEADS + h) * DK) + dk) * S_LEN + s_] =
          (__bf16)acc[r];
    } else {
      ((float*)dstv)[(size_t)m * DMODEL + n] = acc[r];
    }
  }
}

// ---------------------------------------------------------------------------
// Fused attention: scores = Qh@Kh^T /8, mask, exact softmax in LDS,
// stream fp32 weights to d_out, then P@V from the LDS tile.
// Block: 128 threads (4 waves) handles one (b,h) x 16 query rows.
// Dynamic LDS: fp32 score tile 16 x SC_LD + stats (~129 KB, 2 WGs / WGP).
// ---------------------------------------------------------------------------
__global__ void attn_kernel(const __bf16* __restrict__ Qh,
                            const __bf16* __restrict__ Kh,
                            const __bf16* __restrict__ Vt,
                            const unsigned char* __restrict__ mask,
                            float* __restrict__ wOut,
                            __bf16* __restrict__ attnBuf) {
  extern __shared__ float smem[];
  float* sc     = smem;                       // [16][SC_LD]
  float* rowmax = smem + 16 * SC_LD;          // [16]
  float* rowsum = rowmax + 16;                // [16]
  float* red    = rowsum + 16;                // [16][8]

  int tile  = blockIdx.x;
  int qTile = tile & 127;
  int bh    = tile >> 7;
  int h     = bh & 7;
  int b     = bh >> 3;
  int qBase = qTile << 4;

  int lane = threadIdx.x & 31;
  int w    = threadIdx.x >> 5;

  const __bf16* Qp = Qh + ((size_t)bh * S_LEN + qBase) * DK;  // 16 x 64 tile
  const __bf16* Kp = Kh + (size_t)bh * S_LEN * DK;            // 2048 x 64
  const __bf16* Vp = Vt + (size_t)bh * DK * S_LEN;            // 64 x 2048

  // ---- Phase 1: scores tile (16 x 2048) via WMMA, wave w owns 512 keys ----
  {
    bf16x16 a0 = load_a_frag(Qp, DK, 0, 0, lane);
    bf16x16 a1 = load_a_frag(Qp, DK, 0, 32, lane);
    int rbase = (lane & 16) ? 8 : 0;
    for (int t = 0; t < 32; ++t) {
      int nb = w * 512 + t * 16;
      f32x8 acc = {};
      acc = wmma_bf16(a0, load_bt_frag(Kp, DK, nb, 0, lane), acc);
      acc = wmma_bf16(a1, load_bt_frag(Kp, DK, nb, 32, lane), acc);
      int col = nb + (lane & 15);
#pragma unroll
      for (int r = 0; r < 8; ++r) sc[(rbase + r) * SC_LD + col] = acc[r];
    }
  }
  __syncthreads();

  // ---- Phase 2: scale + mask (mask==true -> -1e9, else *0.125) ----
  {
    const unsigned char* mrow = mask + ((size_t)b * S_LEN + qBase) * S_LEN;
    for (int idx = threadIdx.x; idx < 16 * 2048; idx += 128) {
      int row = idx >> 11, col = idx & 2047;
      float v = sc[row * SC_LD + col];
      unsigned char mm = mrow[(size_t)row * S_LEN + col];
      sc[row * SC_LD + col] = mm ? -1.0e9f : v * 0.125f;
    }
  }
  __syncthreads();

  // ---- Phase 3: row max (8 threads / row) ----
  int r   = threadIdx.x >> 3;
  int sub = threadIdx.x & 7;
  {
    float mx = -3.4e38f;
    for (int i = sub; i < 2048; i += 8) mx = fmaxf(mx, sc[r * SC_LD + i]);
    red[r * 8 + sub] = mx;
    __syncthreads();
    if (sub == 0) {
      float m2 = red[r * 8];
#pragma unroll
      for (int j = 1; j < 8; ++j) m2 = fmaxf(m2, red[r * 8 + j]);
      rowmax[r] = m2;
    }
    __syncthreads();
  }

  // ---- Phase 4: exp + row sum ----
  {
    float m2 = rowmax[r];
    float s = 0.0f;
    for (int i = sub; i < 2048; i += 8) {
      float e = __expf(sc[r * SC_LD + i] - m2);
      sc[r * SC_LD + i] = e;
      s += e;
    }
    red[r * 8 + sub] = s;
    __syncthreads();
    if (sub == 0) {
      float t2 = 0.0f;
#pragma unroll
      for (int j = 0; j < 8; ++j) t2 += red[r * 8 + j];
      rowsum[r] = t2;
    }
    __syncthreads();
  }

  // ---- Phase 5: normalize; stream fp32 attention weights (coalesced) ----
  {
    float* wBase = wOut + ((size_t)bh * S_LEN + qBase) * S_LEN;
    for (int idx = threadIdx.x; idx < 16 * 2048; idx += 128) {
      int row = idx >> 11, col = idx & 2047;
      float p = sc[row * SC_LD + col] * (1.0f / rowsum[row]);
      sc[row * SC_LD + col] = p;
      wBase[(size_t)row * S_LEN + col] = p;
    }
  }
  __syncthreads();

  // ---- Phase 6: P @ V  (wave w owns d-columns [w*16, w*16+16)) ----
  {
    int kOff = (lane & 16) ? 8 : 0;
    int m    = lane & 15;
    f32x8 acc = {};
    for (int kb = 0; kb < 2048; kb += 32) {
      bf16x16 a;
#pragma unroll
      for (int j = 0; j < 8; ++j)
        a[j] = (__bf16)sc[m * SC_LD + kb + kOff + j];
#pragma unroll
      for (int j = 0; j < 8; ++j)
        a[8 + j] = (__bf16)sc[m * SC_LD + kb + kOff + 16 + j];
      bf16x16 bf = load_bt_frag(Vp, S_LEN, w * 16, kb, lane);
      acc = wmma_bf16(a, bf, acc);
    }
    int rbase = (lane & 16) ? 8 : 0;
    int n = h * DK + w * 16 + (lane & 15);
#pragma unroll
    for (int rr = 0; rr < 8; ++rr) {
      int mrow = qBase + rbase + rr;
      attnBuf[((size_t)b * S_LEN + mrow) * DMODEL + n] = (__bf16)acc[rr];
    }
  }
}

// ---------------------------------------------------------------------------
extern "C" void kernel_launch(void* const* d_in, const int* in_sizes, int n_in,
                              void* d_out, int out_size, void* d_ws, size_t ws_size,
                              hipStream_t stream) {
  const float* q  = (const float*)d_in[0];
  const float* k  = (const float*)d_in[1];
  const float* v  = (const float*)d_in[2];
  const unsigned char* mask = (const unsigned char*)d_in[3];  // jax bool -> 1B
  const float* w_q = (const float*)d_in[4];
  const float* w_k = (const float*)d_in[5];
  const float* w_v = (const float*)d_in[6];
  const float* w_o = (const float*)d_in[7];

  float* out = (float*)d_out;                               // [4,2048,512]
  float* wts = out + (size_t)BATCH * S_LEN * DMODEL;        // [4,8,2048,2048]

  // workspace carve
  char* ws = (char*)d_ws;
  size_t off = 0;
  auto carve = [&](size_t bytes) -> char* {
    char* p = ws + off;
    off += (bytes + 255) & ~(size_t)255;
    return p;
  };
  const size_t actE = (size_t)MROWS * DMODEL;   // 8192*512
  const size_t wE   = (size_t)DMODEL * DMODEL;  // 512*512
  __bf16* qb  = (__bf16*)carve(actE * 2);
  __bf16* kb  = (__bf16*)carve(actE * 2);
  __bf16* vb  = (__bf16*)carve(actE * 2);
  __bf16* wqb = (__bf16*)carve(wE * 2);
  __bf16* wkb = (__bf16*)carve(wE * 2);
  __bf16* wvb = (__bf16*)carve(wE * 2);
  __bf16* wob = (__bf16*)carve(wE * 2);
  __bf16* Qh  = (__bf16*)carve(actE * 2);   // [B,H,S,DK]
  __bf16* Kh  = (__bf16*)carve(actE * 2);   // [B,H,S,DK]
  __bf16* Vt  = (__bf16*)carve(actE * 2);   // [B,H,DK,S]
  __bf16* aB  = (__bf16*)carve(actE * 2);   // attn concat [B,S,D]

  auto cvt = [&](const float* s, __bf16* d, size_t n) {
    int n8 = (int)(n / 8);
    cvt_f32_bf16<<<(n8 + 255) / 256, 256, 0, stream>>>(s, d, n8);
  };
  cvt(q, qb, actE);   cvt(k, kb, actE);   cvt(v, vb, actE);
  cvt(w_q, wqb, wE);  cvt(w_k, wkb, wE);  cvt(w_v, wvb, wE);  cvt(w_o, wob, wE);

  dim3 pg(MROWS / 16, DMODEL / 64);  // 512 x 8 blocks, 4 waves each
  proj_gemm<<<pg, 128, 0, stream>>>(qb, wqb, (void*)Qh, 0);
  proj_gemm<<<pg, 128, 0, stream>>>(kb, wkb, (void*)Kh, 0);
  proj_gemm<<<pg, 128, 0, stream>>>(vb, wvb, (void*)Vt, 1);

  size_t shmem = (size_t)(16 * SC_LD + 16 + 16 + 128) * sizeof(float); // ~129 KB
  attn_kernel<<<BATCH * NHEADS * (S_LEN / 16), 128, shmem, stream>>>(
      Qh, Kh, Vt, mask, wts, aB);

  proj_gemm<<<pg, 128, 0, stream>>>(aB, wob, (void*)out, 2);
}
